// EG3DTA_62345745269157
// MI455X (gfx1250) — matrix operand order repaired and checked by reading, not compiled, not run on hardware
//
#include <hip/hip_runtime.h>
#include <math.h>

// ===== MI455X / gfx1250: wave32, WMMA f32<-f16 16x16x32 =====
typedef __attribute__((ext_vector_type(16))) _Float16 v16h;
typedef __attribute__((ext_vector_type(8)))  float    v8f;

#define DEVI __device__ __forceinline__

#define NJ  25    // joints
#define TT  100   // frames
#define NMB 4     // N*M
#define NSC 6     // scales
#define NH  4     // heads

// NTU edges (1-based)
__constant__ int c_ea[24] = {1,2,3,4,5,6,7,8,9,10,11,12,13,14,15,16,17,18,19,20,22,23,24,25};
__constant__ int c_eb[24] = {2,21,21,3,21,5,6,7,21,9,10,11,1,13,14,15,1,17,18,19,8,8,12,12};

// ---------------- GEMM functors ----------------
struct AfRM { const float* __restrict__ p; int ld; long zs;
  DEVI float operator()(int r,int k,int z) const { return p[(long)z*zs + (long)r*ld + k]; } };
struct BfRM { const float* __restrict__ p; int ld; long zs;           // B row-major (K x N)
  DEVI float operator()(int k,int c,int z) const { return p[(long)z*zs + (long)k*ld + c]; } };
struct BfT  { const float* __restrict__ p; int ld; long zs;           // B = W^T, W row-major (N x K)
  DEVI float operator()(int k,int c,int z) const { return p[(long)z*zs + (long)c*ld + k]; } };
struct BfSum2T { const float* __restrict__ a; const float* __restrict__ b; int ld;  // (ASC+Ares)^T
  DEVI float operator()(int k,int c,int z) const { long i=(long)c*ld+k; return a[i]+b[i]; } };
struct BfOc { const float* __restrict__ p; int w, V, Vl, T; long zs;  // window-conv gather: K=(c,k)
  DEVI float operator()(int kk,int j,int z) const {
    int c=kk/w, k=kk%w, t=j/V, v=j%V;
    return p[(long)z*zs + ((long)c*T + t)*Vl + k*V + v]; } };

struct CfW { float* __restrict__ p; int ld; long zs;
  DEVI void operator()(int r,int c,int z,float v) const { p[(long)z*zs + (long)r*ld + c] = v; } };
struct CfAdd { float* __restrict__ p; int ld; long zs;
  DEVI void operator()(int r,int c,int z,float v) const { p[(long)z*zs + (long)r*ld + c] += v; } };
struct CfMlpFinal { float* __restrict__ p; int ld; long zs;
  const float* __restrict__ bias; const float* __restrict__ g; const float* __restrict__ b;
  const float* __restrict__ resid; long rzs;
  DEVI void operator()(int r,int c,int z,float v) const {
    long i=(long)z*zs + (long)r*ld + c;
    float t = (p[i] + v + bias[r]) * g[r] + b[r];
    if (resid) t += resid[(long)z*rzs + (long)r*ld + c];
    p[i] = fmaxf(t, 0.f); } };
struct CfOc { float* __restrict__ p; int ld; long zs;
  const float* __restrict__ bias; const float* __restrict__ g; const float* __restrict__ b; int accrelu;
  DEVI void operator()(int r,int c,int z,float v) const {
    long i=(long)z*zs + (long)r*ld + c;
    float val = (v + bias[r]) * g[r] + b[r];
    if (accrelu) p[i] = fmaxf(p[i] + val, 0.f); else p[i] = val; } };
struct CfQkv { float* __restrict__ p; const float* __restrict__ bias; int T, D, H;
  DEVI void operator()(int r,int c,int z,float v) const {      // scatter (bt,o)->(b,h,t,d)
    int bb=r/T, t=r%T, hh=c/D, d=c%D;
    p[(((long)bb*H + hh)*T + t)*D + d] = v + bias[c]; } };
struct CfScale { float* __restrict__ p; int ld; long zs; float s;
  DEVI void operator()(int r,int c,int z,float v) const { p[(long)z*zs + (long)r*ld + c] = v*s; } };
struct CfAv { float* __restrict__ p; int T, D, H, C;            // (z=(b,h),t,d)->(b,t,hD+d)
  DEVI void operator()(int r,int c,int z,float v) const {
    int bb=z/H, hh=z%H;
    p[((long)bb*T + r)*C + hh*D + c] = v; } };
struct CfOut { float* __restrict__ p; const float* __restrict__ xpe; const float* __restrict__ bias;
  int T, V, C;                                                  // +residual,relu -> NCTV
  DEVI void operator()(int r,int c,int z,float v) const {
    int bb=r/T, t=r%T, n=bb/V, vv=bb%V;
    float val = fmaxf(v + bias[c] + xpe[(long)r*C + c], 0.f);
    p[(((long)n*C + c)*T + t)*V + vv] = val; } };

// ---------------- WMMA GEMM core: one wave per 32x32 C macro-tile (2x2 of 16x16) ----------------
// ISA 7.12.2 layouts: A(16x32 f16): row=lane%16, g=lane/16, K(e)= e<8 ? 8g+e : 8+8g+e (+kt)
//                     B(32x16 f16): col=lane%16, K(e)= same as A ;  C(16x16 f32): M=reg+8g, N=lane%16
DEVI v8f wmma_f16(const v16h& a, const v16h& b, const v8f& c) {
  return __builtin_amdgcn_wmma_f32_16x16x32_f16(false, a, false, b, (short)0, c, false, false);
}

template <class AF, class BF, class CF>
__global__ __launch_bounds__(32) void gemm16(AF af, BF bf, CF cf, int M, int N, int K) {
  const int lane = threadIdx.x;           // wave32, block == 1 wave
  const int half = lane >> 4;
  const int i16  = lane & 15;
  const int tm = blockIdx.y * 32, tn = blockIdx.x * 32, z = blockIdx.z;
  const int rowA = tm + i16;              // A rows for this lane (and +16)
  const int colB = tn + i16;              // B/C cols for this lane (and +16)
  v8f acc00 = {0.f,0.f,0.f,0.f,0.f,0.f,0.f,0.f};
  v8f acc01 = acc00, acc10 = acc00, acc11 = acc00;

  const bool interiorMN = (tm + 32 <= M) && (tn + 32 <= N);   // wave-uniform
  // clamped indices for the ragged path (loads always in-range; value selected to 0)
  const int r0c = (rowA     < M) ? rowA     : (M-1);
  const int r1c = (rowA+16  < M) ? rowA+16  : (M-1);
  const int c0c = (colB     < N) ? colB     : (N-1);
  const int c1c = (colB+16  < N) ? colB+16  : (N-1);
  const bool r0ok = rowA < M, r1ok = rowA+16 < M, c0ok = colB < N, c1ok = colB+16 < N;

  for (int kt = 0; kt < K; kt += 32) {
    v16h a0, a1, b0, b1;
    if (interiorMN && (kt + 32 <= K)) {   // fast path: unclamped, k-contiguous -> mergeable loads
#pragma unroll
      for (int e = 0; e < 16; ++e) {
        int ka = kt + ((e < 8) ? (8*half + e) : (8 + 8*half + e));
        int kb = kt + ((e < 8) ? (8*half + e) : (8 + 8*half + e));
        a0[e] = (_Float16)af(rowA,    ka, z);
        a1[e] = (_Float16)af(rowA+16, ka, z);
        b0[e] = (_Float16)bf(kb, colB,    z);
        b1[e] = (_Float16)bf(kb, colB+16, z);
      }
    } else {                              // ragged path: clamp addresses, select values (no branches)
#pragma unroll
      for (int e = 0; e < 16; ++e) {
        int ka = kt + ((e < 8) ? (8*half + e) : (8 + 8*half + e));
        int kb = kt + ((e < 8) ? (8*half + e) : (8 + 8*half + e));
        int kac = (ka < K) ? ka : (K-1);
        int kbc = (kb < K) ? kb : (K-1);
        bool kaok = ka < K, kbok = kb < K;
        float av0 = af(r0c, kac, z);
        float av1 = af(r1c, kac, z);
        float bv0 = bf(kbc, c0c, z);
        float bv1 = bf(kbc, c1c, z);
        a0[e] = (r0ok && kaok) ? (_Float16)av0 : (_Float16)0.f;
        a1[e] = (r1ok && kaok) ? (_Float16)av1 : (_Float16)0.f;
        b0[e] = (c0ok && kbok) ? (_Float16)bv0 : (_Float16)0.f;
        b1[e] = (c1ok && kbok) ? (_Float16)bv1 : (_Float16)0.f;
      }
    }
    acc00 = wmma_f16(a0, b0, acc00);
    acc01 = wmma_f16(a0, b1, acc01);
    acc10 = wmma_f16(a1, b0, acc10);
    acc11 = wmma_f16(a1, b1, acc11);
  }
#pragma unroll
  for (int r = 0; r < 8; ++r) {
    int rr0 = tm + 8*half + r, rr1 = rr0 + 16;
    if (rr0 < M && c0ok) cf(rr0, colB,    z, acc00[r]);
    if (rr0 < M && c1ok) cf(rr0, colB+16, z, acc01[r]);
    if (rr1 < M && c0ok) cf(rr1, colB,    z, acc10[r]);
    if (rr1 < M && c1ok) cf(rr1, colB+16, z, acc11[r]);
  }
}

// ---------------- device-side A_SCALES (graph constants) ----------------
__global__ void k_ascales(int w, float* out, float* scr) {
  const int Vl = w * NJ;
  const int nt = blockDim.x, tid = threadIdx.x;
  float* P  = scr;
  float* Ra = scr + (long)Vl*Vl;
  float* Rb = scr + 2L*Vl*Vl;
  float* dd = scr + 3L*Vl*Vl;
  for (int idx = tid; idx < Vl*Vl; idx += nt) {
    int i = idx/Vl, j = idx%Vl, ii = i%NJ, jj = j%NJ;
    int a = (ii == jj) ? 1 : 0;
    for (int e = 0; e < 24 && !a; ++e)
      if ((c_ea[e]-1==ii && c_eb[e]-1==jj) || (c_eb[e]-1==ii && c_ea[e]-1==jj)) a = 1;
    P[idx]  = (float)a;                 // min(tile(Ab+I)+I, 1)
    Ra[idx] = (i==j) ? 1.f : 0.f;       // reach^0 = I
    out[idx] = (i==j) ? 1.f : 0.f;      // scale 0: norm(I) == I
  }
  __syncthreads();
  float* Rp = Ra; float* Rc = Rb;
  for (int k = 1; k < NSC; ++k) {
    for (int idx = tid; idx < Vl*Vl; idx += nt) {
      int i = idx/Vl, j = idx%Vl;
      float r = 0.f;
      for (int u = 0; u < Vl; ++u)
        if (Rp[i*Vl+u] != 0.f && P[u*Vl+j] != 0.f) { r = 1.f; break; }
      Rc[idx] = r;
    }
    __syncthreads();
    for (int i = tid; i < Vl; i += nt) {
      float s = 0.f;
      for (int j = 0; j < Vl; ++j) s += Rc[i*Vl+j] - Rp[i*Vl+j] + ((i==j)?1.f:0.f);
      dd[i] = (s > 0.f) ? (1.0f/sqrtf(s)) : 0.f;
    }
    __syncthreads();
    for (int idx = tid; idx < Vl*Vl; idx += nt) {
      int i = idx/Vl, j = idx%Vl;
      float m = Rc[idx] - Rp[idx] + ((i==j)?1.f:0.f);
      out[(long)(k*Vl + i)*Vl + j] = m * dd[i] * dd[j];
    }
    __syncthreads();
    float* t = Rp; Rp = Rc; Rc = t;
  }
}

// ---------------- elementwise / reduction glue ----------------
__global__ void k_prep(const float* x, const float* g, const float* b, float* h, int total) {
  for (int i = blockIdx.x*blockDim.x + threadIdx.x; i < total; i += gridDim.x*blockDim.x) {
    int v = i % NJ; int r = i / NJ;
    int t = r % TT; r /= TT;
    int c = r % 3;  int nm = r / 3;
    int m = nm % 2;
    int ch = (m*NJ + v)*3 + c;
    h[i] = x[((long)(nm*TT + t)*NJ + v)*3 + c] * g[ch] + b[ch];
  }
}

__global__ void k_unfold(const float* h, float* xu, int C, int w, int total) {
  int Vl = w*NJ, pad = (w-1)/2;
  for (int i = blockIdx.x*blockDim.x + threadIdx.x; i < total; i += gridDim.x*blockDim.x) {
    int u = i % Vl; int r = i / Vl;
    int t = r % TT; int nc = r / TT;
    int k = u / NJ, v = u % NJ, tt = t + k - pad;
    xu[i] = (tt >= 0 && tt < TT) ? h[((long)nc*TT + tt)*NJ + v] : 0.f;
  }
}

__global__ void k_pe(const float* g, const float* pos, float* xpe, int C, int total) {
  for (int i = blockIdx.x*blockDim.x + threadIdx.x; i < total; i += gridDim.x*blockDim.x) {
    int c = i % C; int r = i / C;
    int t = r % TT; int b = r / TT;
    int n = b / NJ, v = b % NJ;
    xpe[i] = g[(((long)n*C + c)*TT + t)*NJ + v] + pos[t*C + c];
  }
}

__global__ void k_lnbn(const float* xpe, float* hln, const float* lg, const float* lb,
                       const float* bg, const float* bb, int rows, int C) {
  int row = blockIdx.x*(blockDim.x>>5) + (threadIdx.x>>5);
  int lane = threadIdx.x & 31;
  if (row >= rows) return;
  const float* xr = xpe + (long)row*C;
  float s = 0.f, sq = 0.f;
  for (int c = lane; c < C; c += 32) { float x = xr[c]; s += x; sq += x*x; }
  for (int m = 16; m; m >>= 1) { s += __shfl_xor(s, m, 32); sq += __shfl_xor(sq, m, 32); }
  float mean = s / C;
  float var  = sq / C - mean*mean;
  float inv  = rsqrtf(var + 1e-5f);
  float* hr = hln + (long)row*C;
  for (int c = lane; c < C; c += 32) {
    float v = (xr[c] - mean) * inv * lg[c] + lb[c];
    hr[c] = v * bg[c] + bb[c];
  }
}

__global__ void k_softmax(float* sc, int rows, int T) {
  int row = blockIdx.x*(blockDim.x>>5) + (threadIdx.x>>5);
  int lane = threadIdx.x & 31;
  if (row >= rows) return;
  float* r = sc + (long)row*T;
  float mx = -3.4e38f;
  for (int s = lane; s < T; s += 32) mx = fmaxf(mx, r[s]);
  for (int m = 16; m; m >>= 1) mx = fmaxf(mx, __shfl_xor(mx, m, 32));
  float sum = 0.f;
  for (int s = lane; s < T; s += 32) { float e = expf(r[s] - mx); r[s] = e; sum += e; }
  for (int m = 16; m; m >>= 1) sum += __shfl_xor(sum, m, 32);
  float inv = 1.0f / sum;
  for (int s = lane; s < T; s += 32) r[s] *= inv;
}

__global__ void k_pool(const float* h, float* pooled, int nwaves) {
  int wid = blockIdx.x*(blockDim.x>>5) + (threadIdx.x>>5);
  int lane = threadIdx.x & 31;
  if (wid >= nwaves) return;
  const float* base = h + (long)wid * (TT*NJ);
  float s = 0.f;
  for (int j = lane; j < TT*NJ; j += 32) s += base[j];
  for (int m = 16; m; m >>= 1) s += __shfl_xor(s, m, 32);
  if (lane == 0) pooled[wid] = s * (1.0f/(TT*NJ));
}

__global__ void k_fc(const float* pooled, const float* fw, const float* fb, float* out, int Cc) {
  __shared__ float red[256];
  int n = blockIdx.x, t = threadIdx.x;
  float s = 0.f;
  for (int c = t; c < Cc; c += 256)
    s += 0.5f * (pooled[(2*n)*Cc + c] + pooled[(2*n+1)*Cc + c]) * fw[c];
  red[t] = s; __syncthreads();
  for (int o = 128; o > 0; o >>= 1) { if (t < o) red[t] += red[t+o]; __syncthreads(); }
  if (t == 0) out[n] = red[0] + fb[0];
}

// ---------------- workspace layout (floats); needs ws_size >= ~240 MB ----------------
static constexpr long O_SCR = 0;                         // 3*125*125+125
static constexpr long O_A3  = 47104;                     // 450x75
static constexpr long O_A5  = 80896;                     // 750x125
static constexpr long O_H   = 174720;                    // 4*384*100*25
static constexpr long O_G   = O_H   + 3840000;
static constexpr long O_XU  = O_G   + 3840000;           // 4*192*100*125
static constexpr long O_AG  = O_XU  + 9600000;           // agg_s / attention scores
static constexpr long O_H2  = O_AG  + 9600000;
static constexpr long O_XPE = O_H2  + 9600000;
static constexpr long O_HLN = O_XPE + 3840000;
static constexpr long O_Q   = O_HLN + 3840000;
static constexpr long O_K   = O_Q   + 3840000;
static constexpr long O_V   = O_K   + 3840000;
static constexpr long O_O   = O_V   + 3840000;
static constexpr long O_PL  = O_O   + 3840000;           // + 1536

struct G3dP { const float *Ares,*mw,*mb,*bng,*bnb,*ocw,*ocb,*og,*ob; };
struct AttnP { const float *pos,*lng,*lnb,*bng,*bnb,*qw,*qb,*kw,*kb,*vw,*vb,*ow,*obv; };

extern "C" void kernel_launch(void* const* d_in, const int* in_sizes, int n_in,
                              void* d_out, int out_size, void* d_ws, size_t ws_size,
                              hipStream_t stream) {
  (void)in_sizes; (void)n_in; (void)out_size; (void)ws_size;
  int ci = 0;
  auto nx = [&]() { return (const float*)d_in[ci++]; };

  const float* X    = nx();
  const float* dbng = nx();
  const float* dbnb = nx();
  G3dP g3d[3][2]; AttnP at[3];
  for (int bl = 0; bl < 3; ++bl) {
    for (int wi = 0; wi < 2; ++wi) {
      G3dP& p = g3d[bl][wi];
      p.Ares=nx(); p.mw=nx(); p.mb=nx(); p.bng=nx(); p.bnb=nx();
      p.ocw=nx(); p.ocb=nx(); p.og=nx(); p.ob=nx();
    }
    AttnP& a = at[bl];
    a.pos=nx(); a.lng=nx(); a.lnb=nx(); a.bng=nx(); a.bnb=nx();
    a.qw=nx(); a.qb=nx(); a.kw=nx(); a.kb=nx(); a.vw=nx(); a.vb=nx(); a.ow=nx(); a.obv=nx();
  }
  const float* fcw = nx();
  const float* fcb = nx();

  float* ws  = (float*)d_ws;
  float* h   = ws + O_H;
  float* g   = ws + O_G;
  float* xu  = ws + O_XU;
  float* agg = ws + O_AG;
  float* h2  = ws + O_H2;
  float* xpe = ws + O_XPE;
  float* hln = ws + O_HLN;
  float* Qb  = ws + O_Q;
  float* Kb  = ws + O_K;
  float* Vb  = ws + O_V;
  float* Ob  = ws + O_O;
  float* pool = ws + O_PL;

  auto launch = [&](auto af, auto bf, auto cf, int M, int N, int K, int Z) {
    dim3 grid((unsigned)((N+31)/32), (unsigned)((M+31)/32), (unsigned)Z);
    gemm16<<<grid, dim3(32), 0, stream>>>(af, bf, cf, M, N, K);
  };

  // graph constants (device-built: adjacency -> boolean powers -> sym-norm)
  k_ascales<<<1, 256, 0, stream>>>(3, ws + O_A3, ws + O_SCR);
  k_ascales<<<1, 256, 0, stream>>>(5, ws + O_A5, ws + O_SCR);

  // data_bn + layout to (NM,3,T,V)
  {
    int tot = NMB*3*TT*NJ;
    k_prep<<<(tot+255)/256, 256, 0, stream>>>(X, dbng, dbnb, h, tot);
  }

  const int cinA[3]  = {3, 96, 192};
  const int ceoA[3]  = {96, 96, 192};
  const int coutA[3] = {96, 192, 384};

  for (int bl = 0; bl < 3; ++bl) {
    int Cin = cinA[bl], Ceo = ceoA[bl], Cout = coutA[bl];
    for (int wi = 0; wi < 2; ++wi) {
      int w = wi ? 5 : 3, Vl = w*NJ;
      const float* ASC = ws + (wi ? O_A5 : O_A3);
      const G3dP& P = g3d[bl][wi];

      int totU = NMB*Cin*TT*Vl;
      k_unfold<<<(totU+255)/256, 256, 0, stream>>>(h, xu, Cin, w, totU);

      int rows = NMB*Cin*TT, Ncols = TT*Vl;
      for (int s = 0; s < NSC; ++s) {
        // agg_s[(n,c,t), v] = xu[(n,c,t), :] . (ASC+Ares)[sVl+v, :]
        launch(AfRM{xu, Vl, 0},
               BfSum2T{ASC + (long)s*Vl*Vl, P.Ares + (long)s*Vl*Vl, Vl},
               CfW{agg, Vl, 0}, rows, Vl, Vl, 1);
        // h2[n,o,(t,v)] += W[o, s*C + c] . agg_s[n,c,(t,v)]   (z = n)
        AfRM afm{P.mw + s*Cin, NSC*Cin, 0};
        BfRM bfm{agg, Ncols, (long)Cin*Ncols};
        if (s == 0)
          launch(afm, bfm, CfW{h2, Ncols, (long)Ceo*Ncols}, Ceo, Ncols, Cin, NMB);
        else if (s < NSC-1)
          launch(afm, bfm, CfAdd{h2, Ncols, (long)Ceo*Ncols}, Ceo, Ncols, Cin, NMB);
        else
          launch(afm, bfm,
                 CfMlpFinal{h2, Ncols, (long)Ceo*Ncols, P.mb, P.bng, P.bnb,
                            (bl ? xu : nullptr), (long)Cin*Ncols},
                 Ceo, Ncols, Cin, NMB);
      }
      // (1,w,1) conv + obn; wi==1 adds to wi==0 result and applies relu
      launch(AfRM{P.ocw, Ceo*w, 0},
             BfOc{h2, w, NJ, Vl, TT, (long)Ceo*TT*Vl},
             CfOc{g, TT*NJ, (long)Cout*TT*NJ, P.ocb, P.og, P.ob, wi},
             Cout, TT*NJ, Ceo*w, NMB);
    }

    // ---- MHA ----
    const AttnP& A = at[bl];
    int C = Cout, D = C/NH, Brows = NMB*NJ*TT;         // 10000
    int totpe = Brows*C;
    k_pe<<<(totpe+255)/256, 256, 0, stream>>>(g, A.pos, xpe, C, totpe);
    k_lnbn<<<(Brows+7)/8, 256, 0, stream>>>(xpe, hln, A.lng, A.lnb, A.bng, A.bnb, Brows, C);

    launch(AfRM{hln, C, 0}, BfT{A.qw, C, 0}, CfQkv{Qb, A.qb, TT, D, NH}, Brows, C, C, 1);
    launch(AfRM{hln, C, 0}, BfT{A.kw, C, 0}, CfQkv{Kb, A.kb, TT, D, NH}, Brows, C, C, 1);
    launch(AfRM{hln, C, 0}, BfT{A.vw, C, 0}, CfQkv{Vb, A.vb, TT, D, NH}, Brows, C, C, 1);

    int Z = NMB*NJ*NH;                                  // 400 batched attentions
    float* SC = agg;                                    // reuse scratch
    launch(AfRM{Qb, D, (long)TT*D}, BfT{Kb, D, (long)TT*D},
           CfScale{SC, TT, (long)TT*TT, 1.0f/sqrtf((float)D)}, TT, TT, D, Z);
    k_softmax<<<(Z*TT+7)/8, 256, 0, stream>>>(SC, Z*TT, TT);
    launch(AfRM{SC, TT, (long)TT*TT}, BfRM{Vb, D, (long)TT*D},
           CfAv{Ob, TT, D, NH, C}, TT, D, TT, Z);
    launch(AfRM{Ob, C, 0}, BfT{A.ow, C, 0},
           CfOut{h, xpe, A.obv, TT, NJ, C}, Brows, C, C, 1);
  }

  // ---- pool over (T,V), person mean, FC ----
  int Cc = 384;
  k_pool<<<(NMB*Cc+7)/8, 256, 0, stream>>>(h, pool, NMB*Cc);
  k_fc<<<2, 256, 0, stream>>>(pool, fcw, fcb, (float*)d_out, Cc);
}
